// PixelAdaptiveConvolution_13958643712348
// MI455X (gfx1250) — matrix operand order repaired and checked by reading, not compiled, not run on hardware
//
#include <hip/hip_runtime.h>
#include <hip/hip_bf16.h>
#include <stdint.h>

// ---------------------------------------------------------------------------
// PixelAdaptiveConvolution for MI455X (gfx1250), wave32 + WMMA bf16 +
// async-to-LDS double-buffered B staging.
//
// out[p,f] = sum_{c,kl} x[p,c] * patches[p,kl] * w_gen[c, kl*32+f] + bias term.
// GEMM form: A-chunk(ch=c*9+tap)[m,c2] = x[m,c] * x_neighbor(tap)[m,c2]
//            B-chunk(ch)[c2,f]         = ws_bf16[ch*1024 + c2*32 + f]
// Workgroup = 9 waves; wave t async-stages chunk (c+1)*9+t into LDS while all
// waves consume round c's 9 chunks from LDS (double buffered, 80B padded rows
// read by 16-bit LDS loads).
// ---------------------------------------------------------------------------

typedef __attribute__((ext_vector_type(16))) __bf16 v16bf;
typedef __attribute__((ext_vector_type(8)))  float  v8f;
typedef __attribute__((ext_vector_type(4)))  int    v4i;

#define HH 96
#define WW 96
#define CC 32
#define TAPS 9
#define WVB  9                       // waves per block (one per tap)
#define ROW_STRIDE 80                // 64B of B row + 16B pad (bank spread)
#define CHUNK_LDS (32 * ROW_STRIDE)  // 2560 B per 32x32 bf16 chunk
#define ROUND_LDS (TAPS * CHUNK_LDS) // 23040 B per round (9 chunks)
#define LDS_TOTAL (2 * ROUND_LDS)    // 46080 B double buffer

#define AS1 __attribute__((address_space(1)))
#define AS3 __attribute__((address_space(3)))

__device__ __forceinline__ unsigned bf16_1(float f) {
    unsigned u = __float_as_uint(f);
    unsigned r = u + 0x7FFFu + ((u >> 16) & 1u);   // round-nearest-even
    return r >> 16;
}
__device__ __forceinline__ unsigned bfpack(float lo, float hi) {
    return bf16_1(lo) | (bf16_1(hi) << 16);
}
__device__ __forceinline__ unsigned pk_mul_bf16(unsigned a, unsigned b) {
    unsigned d;
    asm("v_pk_mul_bf16 %0, %1, %2" : "=v"(d) : "v"(a), "v"(b));
    return d;
}

__device__ __forceinline__ void async_b128(const void* g, unsigned ldsOff) {
#if __has_builtin(__builtin_amdgcn_global_load_async_to_lds_b128)
    __builtin_amdgcn_global_load_async_to_lds_b128(
        (AS1 v4i*)(uintptr_t)g, (AS3 v4i*)(uintptr_t)ldsOff, 0, 0);
#else
    unsigned long long ga = (unsigned long long)(uintptr_t)g;
    asm volatile("global_load_async_to_lds_b128 %0, %1, off"
                 :: "v"(ldsOff), "v"(ga) : "memory");
#endif
}

__device__ __forceinline__ void wait_async0() {
#if __has_builtin(__builtin_amdgcn_s_wait_asynccnt)
    __builtin_amdgcn_s_wait_asynccnt(0);
#else
    asm volatile("s_wait_asynccnt 0x0" ::: "memory");
#endif
}

union U8toV { unsigned u[8]; v16bf v; };
union Q2toV { uint4 q[2];    v16bf v; };

// --- Pack w_gen (294912 f32) + b_gen (9216 f32) into bf16 workspace ---------
__global__ void pac_pack_bf16(const float* __restrict__ w,
                              const float* __restrict__ b,
                              int nw, unsigned* __restrict__ o, int ntot) {
    int i = blockIdx.x * blockDim.x + threadIdx.x;
    int e = i * 2;
    if (e < ntot) {
        float f0 = (e     < nw) ? w[e]     : b[e - nw];
        float f1 = (e + 1 < nw) ? w[e + 1] : b[e + 1 - nw];
        o[i] = bfpack(f0, f1);
    }
}

// --- Main fused kernel: 9 waves/block, one 16-pixel tile per wave -----------
__global__ __launch_bounds__(WVB * 32) void pac_main(
        const float* __restrict__ x,
        const unsigned char* __restrict__ wsb,   // packed bf16 weights+bias
        float* __restrict__ out,
        int numTiles) {
    __shared__ __align__(16) unsigned char sh[LDS_TOTAL];

    const int t    = threadIdx.x >> 5;           // wave id == staging tap id
    const int lane = threadIdx.x & 31;
    const int tile = blockIdx.x * WVB + t;       // grid sized exactly
    const int p0   = tile * 16;
    const int w0   = p0 % WW;
    const int bh   = p0 / WW;
    const int h    = bh % HH;
    const int b    = bh / HH;
    const int m    = lane & 15;
    const int half = lane >> 4;

    const float* xrow = x + (size_t)(p0 + m) * CC;
    const unsigned shBase = (unsigned)(uintptr_t)(void*)sh;  // LDS offset

    // ---- Preload 9 neighbor taps into bf16 A-operand layout ----------------
    unsigned pt[TAPS][8];
#pragma unroll
    for (int tap = 0; tap < TAPS; ++tap) {
        const int di = tap / 3 - 1, dj = tap % 3 - 1;
        const int h2 = h + di;
        const int w2 = w0 + m + dj;
        float v[16];
#pragma unroll
        for (int j = 0; j < 16; ++j) v[j] = 0.0f;
        if (h2 >= 0 && h2 < HH && w2 >= 0 && w2 < WW) {
            const float* np = x + (((size_t)b * HH + h2) * WW + w2) * CC;
            const float4* qa = (const float4*)(np + half * 8);
            const float4* qb = (const float4*)(np + 16 + half * 8);
            float4 a0 = qa[0], a1 = qa[1], c0 = qb[0], c1 = qb[1];
            v[0] = a0.x; v[1] = a0.y; v[2]  = a0.z; v[3]  = a0.w;
            v[4] = a1.x; v[5] = a1.y; v[6]  = a1.z; v[7]  = a1.w;
            v[8] = c0.x; v[9] = c0.y; v[10] = c0.z; v[11] = c0.w;
            v[12]= c1.x; v[13]= c1.y; v[14] = c1.z; v[15] = c1.w;
        }
#pragma unroll
        for (int i = 0; i < 8; ++i) pt[tap][i] = bfpack(v[2 * i], v[2 * i + 1]);
    }

    // ---- Staging helper: wave t stages chunk r*9+t (2KB = 4 b128/lane) -----
    auto stage = [&](int r, int pr) {
        const unsigned char* g = wsb + (size_t)(r * TAPS + t) * 2048;
        const unsigned lb = shBase + pr * ROUND_LDS + t * CHUNK_LDS;
#pragma unroll
        for (int i = 0; i < 4; ++i) {
            const int e = i * 32 + lane;             // 16B quantum index
            async_b128(g + e * 16, lb + (e >> 2) * ROW_STRIDE + (e & 3) * 16);
        }
    };

    // Prologue: stage round 0, make it visible to the whole block.
    stage(0, 0);
    wait_async0();
    __syncthreads();

    v8f acc0 = {}, acc1 = {};

    // ---- K loop: c = 0..31 real channels, c == 32 is the bias pass ---------
    for (int c = 0; c <= CC; ++c) {
        const int pr = c & 1;
        if (c < CC) stage(c + 1, pr ^ 1);        // overlap next round's DMA

        float xc = (c < CC) ? xrow[c] : 1.0f;
        unsigned sp = bf16_1(xc);
        sp |= sp << 16;                           // replicated bf16 scale

#pragma unroll
        for (int tap = 0; tap < TAPS; ++tap) {
            // B columns from LDS: lane l = column f = l&15 (and 16+f), elements over K rows in table order.
            const unsigned char* cb = sh + pr * ROUND_LDS + tap * CHUNK_LDS + (lane & 15) * 2;

            unsigned av[8];
#pragma unroll
            for (int i = 0; i < 8; ++i) av[i] = pk_mul_bf16(sp, pt[tap][i]);

            U8toV ua;
#pragma unroll
            for (int i = 0; i < 8; ++i) ua.u[i] = av[i];
            U8toV ub0, ub1;
#pragma unroll
            for (int i = 0; i < 8; ++i) {
                const int k = (i < 4 ? 0 : 8) + 8 * half + 2 * i;
                const unsigned short* r0 = (const unsigned short*)(cb + k * ROW_STRIDE);
                const unsigned short* r1 = (const unsigned short*)(cb + (k + 1) * ROW_STRIDE);
                ub0.u[i] = (unsigned)r0[0] | ((unsigned)r1[0] << 16);     // f = 0..15
                ub1.u[i] = (unsigned)r0[16] | ((unsigned)r1[16] << 16);   // f = 16..31
            }

            acc0 = __builtin_amdgcn_wmma_f32_16x16x32_bf16(
                false, ua.v, false, ub0.v, (short)0, acc0, false, false);
            acc1 = __builtin_amdgcn_wmma_f32_16x16x32_bf16(
                false, ua.v, false, ub1.v, (short)0, acc1, false, false);
        }

        // Retire my DMA for round c+1, then block-wide hand-off.
        wait_async0();
        __syncthreads();
    }

    // ---- Store D: vgpr i -> M = i + 8*half, N = lane&15 --------------------
    float* orow = out + (size_t)p0 * CC;
    const int n = lane & 15;
#pragma unroll
    for (int i = 0; i < 8; ++i) {
        const int M = i + 8 * half;
        orow[(size_t)M * CC + n]      = acc0[i];
        orow[(size_t)M * CC + 16 + n] = acc1[i];
    }
}

extern "C" void kernel_launch(void* const* d_in, const int* in_sizes, int n_in,
                              void* d_out, int out_size, void* d_ws, size_t ws_size,
                              hipStream_t stream) {
    const float* x  = (const float*)d_in[0];   // (2,96,96,32) f32
    const float* wg = (const float*)d_in[1];   // (32, 9216)   f32
    const float* bg = (const float*)d_in[2];   // (9216,)      f32

    const int NW   = in_sizes[1];              // 294912
    const int NB   = in_sizes[2];              // 9216
    const int ntot = NW + NB;

    unsigned* ws = (unsigned*)d_ws;            // needs ntot*2 = ~594 KB
    {
        int npair  = ntot / 2;
        int blocks = (npair + 255) / 256;
        pac_pack_bf16<<<blocks, 256, 0, stream>>>(wg, bg, NW, ws, ntot);
    }

    const int numTiles = in_sizes[0] / (CC * 16);   // 1152 tiles of 16 px
    const int blocks   = numTiles / WVB;            // 128 blocks of 9 waves
    pac_main<<<blocks, WVB * 32, 0, stream>>>(x, (const unsigned char*)d_ws,
                                              (float*)d_out, numTiles);
}
